// MyModel_87522843558564
// MI455X (gfx1250) — compile-verified, matrix operand order checked by reading, not run on hardware
//
#include <hip/hip_runtime.h>
#include <hip/hip_bf16.h>

typedef __bf16 bf16_t;
typedef __attribute__((ext_vector_type(16))) __bf16 v16bf;
typedef __attribute__((ext_vector_type(8)))  __bf16 v8bf;
typedef __attribute__((ext_vector_type(8)))  float  v8f;

#define T_STEPS 1024
#define BATCH   32
#define FIN     256
#define UDIM    512
#define XS      264   // 256 + 8 pad (bf16 elems) -> conflict-free ds_load_b128
#define HS      520   // 512 + 8 pad
#define TILE_E  512   // elements per 1KB B tile
#define WSEQ    72    // tiles per wave per direction (24 from W + 48 from U)

static __device__ __forceinline__ bf16_t f2bf(float f) {
  unsigned u = __builtin_bit_cast(unsigned, f);
  unsigned r = u + 0x7FFFu + ((u >> 16) & 1u);   // round-to-nearest-even
  unsigned short h = (unsigned short)(r >> 16);
  return __builtin_bit_cast(bf16_t, h);
}

// A fragment (16x32 bf16, ISA layout): p = rowbase(lane&15) + kbase + (lane>>4)*8
// elements 0..7 = p[0..7], elements 8..15 = p[16..23]  (two 16B ds loads)
static __device__ __forceinline__ v16bf load_a(const bf16_t* p) {
  v8bf lo = *(const v8bf*)p;
  v8bf hi = *(const v8bf*)(p + 16);
  v16bf r;
#pragma unroll
  for (int i = 0; i < 8; ++i) { r[i] = lo[i]; r[i + 8] = hi[i]; }
  return r;
}

// B fragment at sequence position seq from a wave-major packed stream.
// base already includes wave block + lane slot; seq*TILE_E folds into the
// global_load_b128 immediate offset (seq*1024 bytes < 24-bit imm).
static __device__ __forceinline__ v16bf load_b_seq(const bf16_t* base, int seq) {
  return *(const v16bf*)(base + seq * TILE_E);
}

static __device__ __forceinline__ v8f wmma_bf16(v16bf a, v16bf b, v8f c) {
  return __builtin_amdgcn_wmma_f32_16x16x32_bf16(false, a, false, b, (short)0, c,
                                                 false, false);
}

// ---------------------------------------------------------------------------
// K1a: pack GRU weights (W [256x1536], U [512x1536], one direction) into a
// wave-major consumption-ordered bf16 B-tile stream:
//   wave w block = 72 tiles: seq<24 : W, kt=seq/3, gate g=seq%3, nt=w+32g
//                            seq>=24: U, kt=(seq-24)/3, g=(seq-24)%3, nt=w+32g
// Tile = 32 lanes x 16 bf16; lane n16 = col nt*16+n16,
// element e holds K = kt*32 + (e&8)*2 + (lane>>4)*8 + (e&7).
// ---------------------------------------------------------------------------
__global__ void pack_gru_kernel(const float* __restrict__ W, const float* __restrict__ U,
                                bf16_t* __restrict__ dst) {
  int gid = blockIdx.x * blockDim.x + threadIdx.x;
  const int total = 32 * WSEQ * 32;
  if (gid >= total) return;
  int lane = gid & 31;
  int tile = gid >> 5;
  int w    = tile / WSEQ;
  int seq  = tile % WSEQ;
  int n16 = lane & 15, half = (lane >> 4) & 1;
  const float* src;
  int kt, g;
  if (seq < 24) { kt = seq / 3;        g = seq % 3;        src = W; }
  else          { kt = (seq - 24) / 3; g = (seq - 24) % 3; src = U; }
  int col = (w + 32 * g) * 16 + n16;
  bf16_t* out = dst + (((size_t)tile * 32 + lane) << 4);
#pragma unroll
  for (int e = 0; e < 16; ++e) {
    int k = kt * 32 + ((e & 8) << 1) + half * 8 + (e & 7);
    out[e] = f2bf(src[(size_t)k * 1536 + col]);
  }
}

// ---------------------------------------------------------------------------
// K1b: pack Wd [1024 x 50] (cols padded to 64) into per-n-tile contiguous
// streams: tile index = nt*32 + kt  (kt consumed sequentially per wave).
// ---------------------------------------------------------------------------
__global__ void pack_wd_kernel(const float* __restrict__ src, bf16_t* __restrict__ dst) {
  int gid = blockIdx.x * blockDim.x + threadIdx.x;
  const int total = 4 * 32 * 32;
  if (gid >= total) return;
  int lane = gid & 31;
  int tile = gid >> 5;
  int kt = tile & 31, nt = tile >> 5;
  int n16 = lane & 15, half = (lane >> 4) & 1;
  int col = nt * 16 + n16;
  bf16_t* out = dst + (((size_t)tile * 32 + lane) << 4);
#pragma unroll
  for (int e = 0; e < 16; ++e) {
    int k = kt * 32 + ((e & 8) << 1) + half * 8 + (e & 7);
    float v = (col < 50) ? src[(size_t)k * 50 + col] : 0.0f;
    out[e] = f2bf(v);
  }
}

// ---------------------------------------------------------------------------
// K2: persistent GRU scan. One workgroup per direction (blockIdx.x), 32 waves.
// Wave w owns u-tile w: computes N-tiles {w, w+32, w+64} (z / r / h-gate) for
// both 16-row M tiles. h state kept in f32 registers of its owner thread;
// bf16 copy in LDS feeds the next step's WMMA.
// ---------------------------------------------------------------------------
__global__ __launch_bounds__(1024) void gru_scan_kernel(
    const float* __restrict__ x,        // [32][1024][256]
    const float* __restrict__ b_f,      // [2][1536]
    const float* __restrict__ b_b,
    const bf16_t* __restrict__ pAll_f,  // 32 waves * 72 tiles
    const bf16_t* __restrict__ pAll_b,
    bf16_t* __restrict__ h_out)         // [2][1024][32][512]
{
  __shared__ bf16_t xt[BATCH * XS];
  __shared__ bf16_t hbuf[BATCH * HS];

  const int dir = blockIdx.x;
  const float*  bias = dir ? b_b : b_f;
  const bf16_t* pAll = dir ? pAll_b : pAll_f;
  bf16_t* hout = h_out + (size_t)dir * T_STEPS * BATCH * UDIM;

  const int tid  = threadIdx.x;
  const int lane = tid & 31;
  const int wv   = tid >> 5;          // 0..31 : u-tile
  const int n16  = lane & 15;
  const int half = (lane >> 4) & 1;
  const int uc   = wv * 16 + n16;     // this thread's u column (C/D layout)

  // this wave's B-tile stream (wave block + per-lane slot)
  const bf16_t* wavebase = pAll + (size_t)wv * (WSEQ * TILE_E) + lane * 16;

  // biases (z and r fold b_in + b_rec; h-gate keeps them separate)
  const float bz  = bias[uc]        + bias[1536 + uc];
  const float br  = bias[512 + uc]  + bias[1536 + 512 + uc];
  const float bxh = bias[1024 + uc];
  const float brh = bias[1536 + 1024 + uc];

  // zero h (bf16 LDS copy); f32 state lives in registers
  for (int i = tid; i < BATCH * HS; i += 1024) hbuf[i] = f2bf(0.0f);
  float hstate[2][8];
#pragma unroll
  for (int mt = 0; mt < 2; ++mt)
#pragma unroll
    for (int j = 0; j < 8; ++j) hstate[mt][j] = 0.0f;
  __syncthreads();

  for (int step = 0; step < T_STEPS; ++step) {
    const int t = dir ? (T_STEPS - 1 - step) : step;

    // stage x_t -> LDS bf16 (8 elements / thread)
#pragma unroll
    for (int i = 0; i < 8; ++i) {
      int idx = tid + i * 1024;
      int bb_ = idx >> 8;
      int f   = idx & 255;
      xt[bb_ * XS + f] = f2bf(x[((size_t)bb_ * T_STEPS + t) * FIN + f]);
    }
    if (step + 1 < T_STEPS) {             // warm near caches for next step
      int t2 = dir ? (t - 1) : (t + 1);
      int bb_ = tid >> 8, f = tid & 255;
      __builtin_prefetch(&x[((size_t)bb_ * T_STEPS + t2) * FIN + f], 0, 3);
    }
    __syncthreads();                      // x_t visible; prev h writes visible

    float hn[2][8];
#pragma unroll
    for (int mt = 0; mt < 2; ++mt) {
      // Pin the stream base per mt body: blocks LICM over the time loop AND
      // CSE of identical B loads between mt=0 / mt=1 (which would otherwise
      // keep all 72 fragments live -> the spills seen in earlier rounds).
      const bf16_t* wb = wavebase;
      asm volatile("" : "+v"(wb));

      v8f accz, accr, accxh, accrh;
#pragma unroll
      for (int j = 0; j < 8; ++j) { accz[j] = bz; accr[j] = br; accxh[j] = bxh; accrh[j] = brh; }

      const bf16_t* arow_x = xt   + (mt * 16 + n16) * XS + half * 8;
      const bf16_t* arow_h = hbuf + (mt * 16 + n16) * HS + half * 8;

      // input part: K = 256 (8 k-tiles); feeds z, r, xh
#pragma unroll
      for (int kt = 0; kt < 8; ++kt) {
        v16bf a = load_a(arow_x + kt * 32);
        accz  = wmma_bf16(a, load_b_seq(wb, kt * 3 + 0), accz);
        accr  = wmma_bf16(a, load_b_seq(wb, kt * 3 + 1), accr);
        accxh = wmma_bf16(a, load_b_seq(wb, kt * 3 + 2), accxh);
      }
      // recurrent part: K = 512 (16 k-tiles); feeds z, r, rh
#pragma unroll 4
      for (int kt = 0; kt < 16; ++kt) {
        v16bf a = load_a(arow_h + kt * 32);
        accz  = wmma_bf16(a, load_b_seq(wb, 24 + kt * 3 + 0), accz);
        accr  = wmma_bf16(a, load_b_seq(wb, 24 + kt * 3 + 1), accr);
        accrh = wmma_bf16(a, load_b_seq(wb, 24 + kt * 3 + 2), accrh);
      }
      // gates (this thread owns (b = mt*16 + half*8 + j, u = uc))
#pragma unroll
      for (int j = 0; j < 8; ++j) {
        float z   = 1.0f / (1.0f + __expf(-accz[j]));
        float r   = 1.0f / (1.0f + __expf(-accr[j]));
        float pre = accxh[j] + r * accrh[j];
        float e2  = __expf(2.0f * pre);
        float th  = (e2 - 1.0f) / (e2 + 1.0f);
        hn[mt][j] = z * hstate[mt][j] + (1.0f - z) * th;
      }
    }
    __syncthreads();                      // all GEMM reads of hbuf/xt done
#pragma unroll
    for (int mt = 0; mt < 2; ++mt)
#pragma unroll
      for (int j = 0; j < 8; ++j) {
        int brow = mt * 16 + half * 8 + j;
        float v  = hn[mt][j];
        hstate[mt][j] = v;
        bf16_t bv = f2bf(v);
        hbuf[brow * HS + uc] = bv;
        hout[((size_t)t * BATCH + brow) * UDIM + uc] = bv;
      }
    // next iteration's first barrier orders these writes vs. next GEMM reads
  }
}

// ---------------------------------------------------------------------------
// K3: logits = [hf|hb] @ Wd + bd, softmax over 50 classes.
// One block per 16 rows of the flattened [B*T, 1024] A matrix; 4 waves, one
// N-tile (of Wd padded to 64 cols) each; softmax done from an LDS stage.
// ---------------------------------------------------------------------------
__global__ __launch_bounds__(128) void dense_softmax_kernel(
    const bf16_t* __restrict__ h_all,   // [2][1024][32][512]
    const bf16_t* __restrict__ packWd,  // 4*32 tiles (K=1024, Npad=64)
    const float* __restrict__ bd,       // [50]
    float* __restrict__ out)            // [32][1024][50]
{
  __shared__ float logits[16][64];
  const int mt   = blockIdx.x;          // 0..2047
  const int tid  = threadIdx.x;
  const int lane = tid & 31;
  const int wv   = tid >> 5;            // n-tile 0..3
  const int n16  = lane & 15;
  const int half = (lane >> 4) & 1;

  // A-row for this lane
  const int grow = mt * 16 + n16;
  const int bidx = grow >> 10;
  const int tt   = grow & 1023;
  const bf16_t* hf_row = h_all + ((size_t)tt * BATCH + bidx) * UDIM;
  const bf16_t* hb_row = hf_row + (size_t)T_STEPS * BATCH * UDIM;

  // this wave's Wd tile stream
  const bf16_t* wdb = packWd + (size_t)wv * 32 * TILE_E + lane * 16;
  asm volatile("" : "+v"(wdb));

  v8f acc;
#pragma unroll
  for (int j = 0; j < 8; ++j) acc[j] = 0.0f;

#pragma unroll 4
  for (int kt = 0; kt < 32; ++kt) {
    const bf16_t* src = (kt < 16) ? (hf_row + kt * 32) : (hb_row + (kt - 16) * 32);
    v16bf a = load_a(src + half * 8);
    acc = wmma_bf16(a, load_b_seq(wdb, kt), acc);
  }
#pragma unroll
  for (int j = 0; j < 8; ++j)
    logits[half * 8 + j][wv * 16 + n16] = acc[j];
  __syncthreads();

  if (tid < 16) {
    const int lr = tid;
    const int g2 = mt * 16 + lr;
    const int b2 = g2 >> 10, t2 = g2 & 1023;
    float mx = -3.4e38f;
    for (int c = 0; c < 50; ++c) {
      float v = logits[lr][c] + bd[c];
      logits[lr][c] = v;
      mx = v > mx ? v : mx;
    }
    float s = 0.0f;
    for (int c = 0; c < 50; ++c) {
      float e = __expf(logits[lr][c] - mx);
      logits[lr][c] = e;
      s += e;
    }
    float inv = 1.0f / s;
    float* o = out + ((size_t)b2 * T_STEPS + t2) * 50;
    for (int c = 0; c < 50; ++c) o[c] = logits[lr][c] * inv;
  }
}

// ---------------------------------------------------------------------------
extern "C" void kernel_launch(void* const* d_in, const int* in_sizes, int n_in,
                              void* d_out, int out_size, void* d_ws, size_t ws_size,
                              hipStream_t stream) {
  (void)in_sizes; (void)n_in; (void)out_size; (void)ws_size;
  const float* x   = (const float*)d_in[0];
  const float* Wf  = (const float*)d_in[1];
  const float* Uf  = (const float*)d_in[2];
  const float* bf_ = (const float*)d_in[3];
  const float* Wb  = (const float*)d_in[4];
  const float* Ub  = (const float*)d_in[5];
  const float* bb_ = (const float*)d_in[6];
  const float* Wd  = (const float*)d_in[7];
  const float* bd  = (const float*)d_in[8];
  float* out = (float*)d_out;

  char* ws = (char*)d_ws;
  bf16_t* pAllf = (bf16_t*)(ws + 0);          // 32*72 tiles * 1KB = 2359296 B
  bf16_t* pAllb = (bf16_t*)(ws + 2359296);    // 2359296 B
  bf16_t* pWd   = (bf16_t*)(ws + 4718592);    // 128 tiles = 131072 B
  bf16_t* hAll  = (bf16_t*)(ws + 4849664);    // 64 MB bf16 hidden states

  pack_gru_kernel<<<288, 256, 0, stream>>>(Wf, Uf, pAllf);   // 73728 threads
  pack_gru_kernel<<<288, 256, 0, stream>>>(Wb, Ub, pAllb);
  pack_wd_kernel<<<16, 256, 0, stream>>>(Wd, pWd);

  gru_scan_kernel<<<2, 1024, 0, stream>>>(x, bf_, bb_, pAllf, pAllb, hAll);
  dense_softmax_kernel<<<2048, 128, 0, stream>>>(hAll, pWd, bd, out);
}
